// ColorHistCriterion_56521769615944
// MI455X (gfx1250) — compile-verified, matrix operand order checked by reading, not run on hardware
//
#include <hip/hip_runtime.h>

#define NBINS   255
#define HW      262144      // 512*512 pixels per (batch, channel)
#define NBC     96          // 32 batches * 3 channels
#define SLICES  8           // blocks per (b,c) in the binning pass
#define NTOT    (NBC * NBINS)

typedef __attribute__((ext_vector_type(16))) _Float16 v16h;
typedef __attribute__((ext_vector_type(8)))  float    v8f;

// ---------------------------------------------------------------------------
// Pass 1: per-(b,c) min/max of round(x*255).  One block per (b,c),
// float4 (global_load_b128) streaming, LDS tree reduction.
// ---------------------------------------------------------------------------
__global__ __launch_bounds__(256) void k_minmax(const float* __restrict__ x,
                                                float* __restrict__ lohi) {
  const int bc = blockIdx.x;
  const float4* p = (const float4*)(x + (size_t)bc * HW);
  float vmin =  1e30f;
  float vmax = -1e30f;
  for (int i = threadIdx.x; i < HW / 4; i += 256) {
    float4 f = p[i];
    float a = rintf(f.x * 255.0f);
    float b = rintf(f.y * 255.0f);
    float c = rintf(f.z * 255.0f);
    float d = rintf(f.w * 255.0f);
    vmin = fminf(vmin, fminf(fminf(a, b), fminf(c, d)));
    vmax = fmaxf(vmax, fmaxf(fmaxf(a, b), fmaxf(c, d)));
  }
  __shared__ float smin[256];
  __shared__ float smax[256];
  smin[threadIdx.x] = vmin;
  smax[threadIdx.x] = vmax;
  __syncthreads();
  for (int s = 128; s > 0; s >>= 1) {
    if (threadIdx.x < s) {
      smin[threadIdx.x] = fminf(smin[threadIdx.x], smin[threadIdx.x + s]);
      smax[threadIdx.x] = fmaxf(smax[threadIdx.x], smax[threadIdx.x + s]);
    }
    __syncthreads();
  }
  if (threadIdx.x == 0) {
    lohi[bc * 2 + 0] = smin[0];
    lohi[bc * 2 + 1] = smax[0];
  }
}

// ---------------------------------------------------------------------------
// Zero the global histogram accumulator in workspace.
// ---------------------------------------------------------------------------
__global__ __launch_bounds__(256) void k_zero(float* __restrict__ h, int n) {
  int i = blockIdx.x * 256 + threadIdx.x;
  if (i < n) h[i] = 0.0f;
}

// ---------------------------------------------------------------------------
// Pass 2: binning.  SLICES blocks per (b,c); each block privatizes the 255
// bins in LDS (ds_add_u32 atomics), then flushes with global_atomic_add_f32.
// Exact: all counts are integers < 2^24.
// ---------------------------------------------------------------------------
__global__ __launch_bounds__(256) void k_hist(const float* __restrict__ x,
                                              const float* __restrict__ lohi,
                                              float* __restrict__ hist) {
  const int bc    = blockIdx.x / SLICES;
  const int slice = blockIdx.x % SLICES;
  const float lo    = lohi[bc * 2 + 0];
  const float hi    = lohi[bc * 2 + 1];
  const float width = (hi > lo) ? (hi - lo) : 1.0f;
  const float scale = (float)NBINS / width;

  __shared__ unsigned bins[NBINS];
  for (int i = threadIdx.x; i < NBINS; i += 256) bins[i] = 0u;
  __syncthreads();

  const int per = HW / SLICES;   // 32768 pixels per slice
  const float4* p = (const float4*)(x + (size_t)bc * HW + (size_t)slice * per);
  for (int i = threadIdx.x; i < per / 4; i += 256) {
    float4 f = p[i];
    float v[4] = {f.x, f.y, f.z, f.w};
#pragma unroll
    for (int j = 0; j < 4; ++j) {
      float vv  = rintf(v[j] * 255.0f);
      int   idx = (int)floorf((vv - lo) * scale);
      idx = idx < 0 ? 0 : (idx > NBINS - 1 ? NBINS - 1 : idx);
      atomicAdd(&bins[idx], 1u);                 // ds_add_u32
    }
  }
  __syncthreads();

  for (int i = threadIdx.x; i < NBINS; i += 256) {
    unsigned c = bins[i];
    if (c) atomicAdd(&hist[bc * NBINS + i], (float)c);  // global_atomic_add_f32
  }
}

// ---------------------------------------------------------------------------
// Pass 3: mean(|h - h|) via the matrix unit.  One wave32.  Per iteration each
// lane packs 16 diffs (f16, exact for 0) into the A operand and multiplies by
// an all-ones B: D = A*1 accumulates row sums in C.  Placement inside A is
// irrelevant for a total sum.  Final total = column-0 of D = lane0 partial
// (rows 0-7) + lane16 partial (rows 8-15).
// ---------------------------------------------------------------------------
__global__ __launch_bounds__(32) void k_reduce(const float* __restrict__ hist,
                                               float* __restrict__ out, int n) {
  const int lane = threadIdx.x;

  v16h ones;
#pragma unroll
  for (int i = 0; i < 16; ++i) ones[i] = (_Float16)1.0f;

  v8f acc = {};
  for (int base = 0; base < n; base += 512) {   // 32 lanes * 16 values
    v16h a;
#pragma unroll
    for (int i = 0; i < 16; ++i) {
      int   idx  = base + lane * 16 + i;
      int   cidx = idx < n ? idx : 0;           // clamp: branch-free, EXEC stays all-1s
      float h    = hist[cidx];
      float d    = fabsf(h - h);
      a[i] = (_Float16)(idx < n ? d : 0.0f);
    }
    // 8 args: (neg_a, A, neg_b, B, c_mod, C, reuse_a, reuse_b)
    acc = __builtin_amdgcn_wmma_f32_16x16x32_f16(
        /*neg_a=*/false, a, /*neg_b=*/false, ones,
        /*c_mod=*/(short)0, acc, /*reuse_a=*/false, /*reuse_b=*/false);
  }

  float partial = acc[0] + acc[1] + acc[2] + acc[3] +
                  acc[4] + acc[5] + acc[6] + acc[7];
  float upper = __shfl(partial, 16, 32);        // rows 8..15 of column 0
  if (lane == 0) out[0] = (partial + upper) / (float)n;
}

// ---------------------------------------------------------------------------
extern "C" void kernel_launch(void* const* d_in, const int* in_sizes, int n_in,
                              void* d_out, int out_size, void* d_ws, size_t ws_size,
                              hipStream_t stream) {
  (void)in_sizes; (void)n_in; (void)out_size; (void)ws_size;
  const float* x = (const float*)d_in[0];       // (32,3,512,512) f32; y unused
  float* lohi = (float*)d_ws;                   // 192 floats
  float* hist = lohi + 2 * NBC;                 // 24480 floats
  float* out  = (float*)d_out;

  k_minmax<<<NBC, 256, 0, stream>>>(x, lohi);
  k_zero  <<<(NTOT + 255) / 256, 256, 0, stream>>>(hist, NTOT);
  k_hist  <<<NBC * SLICES, 256, 0, stream>>>(x, lohi, hist);
  k_reduce<<<1, 32, 0, stream>>>(hist, out, NTOT);
}
